// Attention_17781164605679
// MI455X (gfx1250) — matrix operand order repaired and checked by reading, not compiled, not run on hardware
//
#include <hip/hip_runtime.h>

// ---------------------------------------------------------------------------
// Problem constants (from reference): B=2, S=2048, H=2048, NH=16, NKV=4, HD=128
// ---------------------------------------------------------------------------
#define CB   2
#define CS   2048
#define CH   2048
#define CNH  16
#define CNKV 4
#define CHD  128
#define CNREP (CNH / CNKV)

typedef __attribute__((ext_vector_type(8)))  __bf16 v8bf;
typedef __attribute__((ext_vector_type(16))) __bf16 v16bf;
typedef __attribute__((ext_vector_type(8)))  float  v8f;
typedef __attribute__((ext_vector_type(4)))  int    v4i;

#if __has_builtin(__builtin_amdgcn_global_load_async_to_lds_b128) && \
    __has_builtin(__builtin_amdgcn_s_wait_asynccnt)
#define HAVE_ASYNC_LDS 1
#else
#define HAVE_ASYNC_LDS 0
#endif

typedef __attribute__((address_space(1))) v4i g_v4i;   // global AS pointer-ee
typedef __attribute__((address_space(3))) v4i l_v4i;   // LDS AS pointer-ee

__device__ __forceinline__ v8f v8f_zero() {
    v8f z = {0.f, 0.f, 0.f, 0.f, 0.f, 0.f, 0.f, 0.f};
    return z;
}

// 16-byte global -> LDS copy; async (ASYNCcnt-tracked, no VGPR staging) when
// the gfx1250 builtin is available, otherwise a plain load+store.
__device__ __forceinline__ void copy16_to_lds(const __bf16* g, __bf16* l) {
#if HAVE_ASYNC_LDS
    __builtin_amdgcn_global_load_async_to_lds_b128(
        (g_v4i*)(uintptr_t)g,
        (l_v4i*)(uintptr_t)(uint32_t)(uintptr_t)l,
        0, 0);
#else
    *(int4*)l = *(const int4*)g;
#endif
}

__device__ __forceinline__ void wait_async_keep6() {
#if HAVE_ASYNC_LDS
    __builtin_amdgcn_s_wait_asynccnt(6);   // let next tile's 6 copies fly
#endif
}
__device__ __forceinline__ void wait_async_all() {
#if HAVE_ASYNC_LDS
    __builtin_amdgcn_s_wait_asynccnt(0);
#endif
}

// A-matrix fragment (16x32 bf16, M x K). Row-major source, row stride = ldk.
// ISA layout: lanes 0-15 -> M=lane, K = 0..7 then 16..23 (per half of v16bf);
// lanes 16-31 -> same M, K = 8..15 then 24..31.
__device__ __forceinline__ v16bf load_afrag(const __bf16* __restrict__ A,
                                            int ldk, int row0, int k0) {
    const int lane = threadIdx.x & 31;
    const int lo = lane & 15;
    const int hi = lane >> 4;
    const __bf16* p = A + (size_t)(row0 + lo) * ldk + (k0 + hi * 8);
    v8bf a0 = *(const v8bf*)p;          // K = base .. base+7
    v8bf a1 = *(const v8bf*)(p + 16);   // K = base+16 .. base+23
    return __builtin_shufflevector(a0, a1, 0, 1, 2, 3, 4, 5, 6, 7,
                                           8, 9, 10, 11, 12, 13, 14, 15);
}

// B-matrix fragment (32x16 bf16, K x N) loaded from a transposed [N,K] source:
// lanes 0-15 -> N=lane, K=0..7 then 16..23; lanes 16-31 -> same N, K=8..15 then 24..31.
// Two 16B loads so only 16B alignment is required (LDS tiles use padded rows).
__device__ __forceinline__ v16bf load_bfrag(const __bf16* __restrict__ Bt,
                                            int ldk, int col0, int k0) {
    const int lane = threadIdx.x & 31;
    const int lo = lane & 15;
    const int hi = lane >> 4;
    const __bf16* p = Bt + (size_t)(col0 + lo) * ldk + (k0 + hi * 8);
    v8bf b0 = *(const v8bf*)p;
    v8bf b1 = *(const v8bf*)(p + 16);
    return __builtin_shufflevector(b0, b1, 0, 1, 2, 3, 4, 5, 6, 7,
                                           8, 9, 10, 11, 12, 13, 14, 15);
}

__device__ __forceinline__ v8f wmma_bf16(v16bf a, v16bf b, v8f c) {
    return __builtin_amdgcn_wmma_f32_16x16x32_bf16(
        /*neg_a=*/false, a, /*neg_b=*/false, b,
        /*c_mod=*/(short)0, c, /*reuse_a=*/false, /*reuse_b=*/false);
}

// ---------------------------------------------------------------------------
// fp32 -> bf16 elementwise convert
// ---------------------------------------------------------------------------
__global__ void cvt_f32_bf16(const float* __restrict__ src,
                             __bf16* __restrict__ dst, size_t n) {
    size_t i = (size_t)blockIdx.x * blockDim.x + threadIdx.x;
    if (i < n) dst[i] = (__bf16)src[i];
}

// fp32 [K,N] row-major -> bf16 [N,K] row-major (weight transpose for WMMA B)
__global__ void transpose_cvt(const float* __restrict__ w,
                              __bf16* __restrict__ wt, int K, int N) {
    size_t idx = (size_t)blockIdx.x * blockDim.x + threadIdx.x;
    if (idx >= (size_t)K * N) return;
    int k = (int)(idx % K);
    size_t nrow = idx / K;
    wt[idx] = (__bf16)w[(size_t)k * N + nrow];
}

// ---------------------------------------------------------------------------
// bf16 WMMA GEMM:  C[M,N] = A[M,K] * Bt[N,K]^T
// Block tile 64(M) x 128(N) x 64(K), double-buffered async-copy into LDS.
// 256 threads = 8 waves as 4(M) x 2(N); wave tile = 16 x 64.
// Per k-tile each thread issues 6 async 16B copies (A:2, B:4).
// ---------------------------------------------------------------------------
#define GBM 64
#define GBN 128
#define GBK 64
#define GLDT (GBK + 8)      // padded LDS row stride (bank-spread: 36 dwords)

template <bool OUT_F32>
__global__ __launch_bounds__(256)
void gemm_bf16_wmma(const __bf16* __restrict__ A, const __bf16* __restrict__ Bt,
                    float* __restrict__ Cf, __bf16* __restrict__ Cb,
                    int M, int N, int K) {
    __shared__ __bf16 As[2][GBM * GLDT];
    __shared__ __bf16 Bs[2][GBN * GLDT];

    const int tid  = threadIdx.x;
    const int wave = tid >> 5;
    const int lane = tid & 31;
    const int lo = lane & 15;
    const int hi = lane >> 4;
    const int wm = wave & 3;            // 0..3 -> M sub-tile
    const int wn = wave >> 2;           // 0..1 -> N sub-tile
    const int rowg = blockIdx.y * GBM;
    const int colg = blockIdx.x * GBN;

    auto stage = [&](int buf, int k0) {
        // A tile: 64 rows x 64 k = 512 x 16B chunks, 2 per thread
#pragma unroll
        for (int i = 0; i < 2; ++i) {
            int c = tid + 256 * i;
            int r = c >> 3, cc = c & 7;
            copy16_to_lds(A + (size_t)(rowg + r) * K + k0 + cc * 8,
                          &As[buf][r * GLDT + cc * 8]);
        }
        // B tile: 128 rows x 64 k = 1024 x 16B chunks, 4 per thread
#pragma unroll
        for (int i = 0; i < 4; ++i) {
            int c = tid + 256 * i;
            int r = c >> 3, cc = c & 7;
            copy16_to_lds(Bt + (size_t)(colg + r) * K + k0 + cc * 8,
                          &Bs[buf][r * GLDT + cc * 8]);
        }
    };

    v8f acc0 = v8f_zero(), acc1 = v8f_zero(), acc2 = v8f_zero(), acc3 = v8f_zero();

    const int nt = K / GBK;
    stage(0, 0);
    for (int kt = 0; kt < nt; ++kt) {
        const int buf = kt & 1;
        if (kt + 1 < nt) {
            stage(buf ^ 1, (kt + 1) * GBK);
            wait_async_keep6();         // current tile's copies retired
        } else {
            wait_async_all();
        }
        __syncthreads();
#pragma unroll
        for (int ks = 0; ks < GBK; ks += 32) {
            v16bf a  = load_afrag(&As[buf][0], GLDT, wm * 16, ks);
            v16bf b0 = load_bfrag(&Bs[buf][0], GLDT, wn * 64 + 0,  ks);
            v16bf b1 = load_bfrag(&Bs[buf][0], GLDT, wn * 64 + 16, ks);
            v16bf b2 = load_bfrag(&Bs[buf][0], GLDT, wn * 64 + 32, ks);
            v16bf b3 = load_bfrag(&Bs[buf][0], GLDT, wn * 64 + 48, ks);
            acc0 = wmma_bf16(a, b0, acc0);
            acc1 = wmma_bf16(a, b1, acc1);
            acc2 = wmma_bf16(a, b2, acc2);
            acc3 = wmma_bf16(a, b3, acc3);
        }
        __syncthreads();
    }

    const int row0 = rowg + wm * 16;
    const int col0 = colg + wn * 64;
    v8f accs[4] = {acc0, acc1, acc2, acc3};
#pragma unroll
    for (int t = 0; t < 4; ++t) {
#pragma unroll
        for (int r = 0; r < 8; ++r) {
            int row = row0 + r + 8 * hi;
            int col = col0 + t * 16 + lo;
            if (OUT_F32) Cf[(size_t)row * N + col] = accs[t][r];
            else         Cb[(size_t)row * N + col] = (__bf16)accs[t][r];
        }
    }
}

// ---------------------------------------------------------------------------
// RoPE + scatter: raw [B*S, nheads*HD] bf16 -> out [B, nheads, S, HD] bf16
// ---------------------------------------------------------------------------
__global__ void rope_scatter(const __bf16* __restrict__ raw,
                             const float* __restrict__ cosb,
                             const float* __restrict__ sinb,
                             __bf16* __restrict__ out, int nheads) {
    size_t idx = (size_t)blockIdx.x * blockDim.x + threadIdx.x;
    int d = (int)(idx % CHD);
    size_t t = idx / CHD;
    int h = (int)(t % nheads);
    size_t bs = t / nheads;
    int s = (int)(bs % CS);
    int b = (int)(bs / CS);
    if (b >= CB) return;

    size_t base = bs * (size_t)(nheads * CHD) + (size_t)h * CHD;
    float x0 = (float)raw[base + d];
    float xr = (d < CHD / 2) ? -(float)raw[base + d + CHD / 2]
                             :  (float)raw[base + d - CHD / 2];
    float c  = cosb[(size_t)s * CHD + d];
    float sn = sinb[(size_t)s * CHD + d];
    out[(((size_t)b * nheads + h) * CS + s) * CHD + d] = (__bf16)(x0 * c + xr * sn);
}

// V scatter transposed: raw [B*S, NKV*HD] -> vt [B, NKV, HD, S]
__global__ void v_scatter_t(const __bf16* __restrict__ raw,
                            __bf16* __restrict__ vt) {
    size_t idx = (size_t)blockIdx.x * blockDim.x + threadIdx.x;
    int d = (int)(idx % CHD);
    size_t t = idx / CHD;
    int h = (int)(t % CNKV);
    size_t bs = t / CNKV;
    int s = (int)(bs % CS);
    int b = (int)(bs / CS);
    if (b >= CB) return;
    vt[(((size_t)b * CNKV + h) * CHD + d) * CS + s] =
        raw[bs * (size_t)(CNKV * CHD) + (size_t)h * CHD + d];
}

// ---------------------------------------------------------------------------
// Flash attention: one wave (32 threads) per (b, head, 16-query tile).
// q [B,NH,S,HD], k [B,NKV,S,HD], vt [B,NKV,HD,S]  (all bf16, RoPE applied)
// out: attn_out [B, S, NH*HD] bf16
// ---------------------------------------------------------------------------
__global__ __launch_bounds__(32)
void flash_attn(const __bf16* __restrict__ q, const __bf16* __restrict__ k,
                const __bf16* __restrict__ vt, __bf16* __restrict__ attn_out) {
    const int qt = blockIdx.x;          // query tile (16 rows)
    const int h  = blockIdx.y;          // head
    const int b  = blockIdx.z;          // batch
    const int kvh = h / CNREP;          // GQA: contiguous repeat

    const __bf16* qb = q  + ((size_t)b * CNH  + h)   * CS * CHD;
    const __bf16* kb = k  + ((size_t)b * CNKV + kvh) * CS * CHD;
    const __bf16* vb = vt + ((size_t)b * CNKV + kvh) * CHD * CS;

    const int q0 = qt * 16;
    const int lane = threadIdx.x & 31;
    const int lo = lane & 15;
    const int hi = lane >> 4;
    const float scale = 0.08838834764831845f;   // 1/sqrt(128)

    // Resident Q fragments over HD (4 chunks of K=32)
    v16bf qa[4];
#pragma unroll
    for (int c = 0; c < 4; ++c) qa[c] = load_afrag(qb, CHD, q0, c * 32);

    v8f acc[8];
#pragma unroll
    for (int t = 0; t < 8; ++t) acc[t] = v8f_zero();
    float m[8], l[8];
#pragma unroll
    for (int r = 0; r < 8; ++r) { m[r] = -1e30f; l[r] = 0.f; }

    __shared__ float pbuf[16 * 32];     // P staging: C-layout -> A-layout

    const int kend = q0 + 16;           // causal: keys <= q0+15
    for (int kc = 0; kc < kend; kc += 32) {
        // ---- scores S = Q K^T for 2 x (16q x 16key) tiles, K-dim = HD ----
        v8f s0 = v8f_zero(), s1 = v8f_zero();
#pragma unroll
        for (int c = 0; c < 4; ++c) {
            v16bf bk0 = load_bfrag(kb, CHD, kc,      c * 32);
            v16bf bk1 = load_bfrag(kb, CHD, kc + 16, c * 32);
            s0 = wmma_bf16(qa[c], bk0, s0);
            s1 = wmma_bf16(qa[c], bk1, s1);
        }
        // ---- causal mask + scale + row max (rows r + 8*hi across 16 lanes) --
        float rowmax[8];
#pragma unroll
        for (int r = 0; r < 8; ++r) {
            int row = q0 + r + 8 * hi;
            float v0 = s0[r] * scale;
            float v1 = s1[r] * scale;
            v0 = ((kc + lo)      > row) ? -1e30f : v0;
            v1 = ((kc + 16 + lo) > row) ? -1e30f : v1;
            s0[r] = v0; s1[r] = v1;
            float x = fmaxf(v0, v1);
#pragma unroll
            for (int off = 1; off < 16; off <<= 1)
                x = fmaxf(x, __shfl_xor(x, off, 16));
            rowmax[r] = x;
        }
        // ---- online softmax update ----
        float alpha[8];
#pragma unroll
        for (int r = 0; r < 8; ++r) {
            float mn = fmaxf(m[r], rowmax[r]);
            alpha[r] = __expf(m[r] - mn);
            m[r] = mn;
            float p0 = __expf(s0[r] - mn);
            float p1 = __expf(s1[r] - mn);
            s0[r] = p0; s1[r] = p1;
            float ps = p0 + p1;
#pragma unroll
            for (int off = 1; off < 16; off <<= 1)
                ps += __shfl_xor(ps, off, 16);
            l[r] = l[r] * alpha[r] + ps;
        }
        // ---- re-layout P (C frag -> A frag) via LDS ----
#pragma unroll
        for (int r = 0; r < 8; ++r) {
            int row = r + 8 * hi;
            pbuf[row * 32 + lo]      = s0[r];
            pbuf[row * 32 + 16 + lo] = s1[r];
        }
        __syncthreads();
        v16bf pa;
#pragma unroll
        for (int i = 0; i < 8; ++i) {
            pa[i]     = (__bf16)pbuf[lo * 32 + hi * 8 + i];
            pa[i + 8] = (__bf16)pbuf[lo * 32 + hi * 8 + 16 + i];
        }
        __syncthreads();
        // ---- rescale accumulators, accumulate P V (K-dim = 32 keys) ----
#pragma unroll
        for (int t = 0; t < 8; ++t) {
#pragma unroll
            for (int r = 0; r < 8; ++r) acc[t][r] *= alpha[r];
            v16bf bv = load_bfrag(vb, CS, t * 16, kc);  // vt: [HD, S]
            acc[t] = wmma_bf16(pa, bv, acc[t]);
        }
    }

    // ---- epilogue: divide by l, write [B, S, NH*HD] ----
    __bf16* ob = attn_out + (size_t)b * CS * (CNH * CHD) + (size_t)h * CHD;
#pragma unroll
    for (int t = 0; t < 8; ++t) {
#pragma unroll
        for (int r = 0; r < 8; ++r) {
            int row = q0 + r + 8 * hi;
            ob[(size_t)row * (CNH * CHD) + t * 16 + lo] = (__bf16)(acc[t][r] / l[r]);
        }
    }
}

// ---------------------------------------------------------------------------
// Host-side orchestration
// ---------------------------------------------------------------------------
extern "C" void kernel_launch(void* const* d_in, const int* in_sizes, int n_in,
                              void* d_out, int out_size, void* d_ws, size_t ws_size,
                              hipStream_t stream) {
    const float* x    = (const float*)d_in[0];
    const float* cosb = (const float*)d_in[1];
    const float* sinb = (const float*)d_in[2];
    const float* wq   = (const float*)d_in[3];
    const float* wk   = (const float*)d_in[4];
    const float* wv   = (const float*)d_in[5];
    const float* wo   = (const float*)d_in[6];
    float* out = (float*)d_out;

    char* ws = (char*)d_ws;
    size_t off = 0;
    auto take = [&](size_t elems) -> __bf16* {
        __bf16* p = (__bf16*)(ws + off);
        off += ((elems * sizeof(__bf16)) + 255) & ~(size_t)255;
        return p;
    };

    const size_t MROWS = (size_t)CB * CS;             // 4096
    __bf16* xb   = take(MROWS * CH);                  // x in bf16
    __bf16* wqT  = take((size_t)CH * (CNH * CHD));    // [N=2048, K=2048]
    __bf16* wkT  = take((size_t)CH * (CNKV * CHD));   // [512, 2048]
    __bf16* wvT  = take((size_t)CH * (CNKV * CHD));   // [512, 2048]
    __bf16* woT  = take((size_t)(CNH * CHD) * CH);    // [2048, 2048]
    __bf16* qraw = take(MROWS * (CNH * CHD));
    __bf16* kraw = take(MROWS * (CNKV * CHD));
    __bf16* vraw = take(MROWS * (CNKV * CHD));
    __bf16* qr   = take(MROWS * (CNH * CHD));         // [B,NH,S,HD]
    __bf16* kr   = take(MROWS * (CNKV * CHD));        // [B,NKV,S,HD]
    __bf16* vtb  = take(MROWS * (CNKV * CHD));        // [B,NKV,HD,S]
    __bf16* ao   = take(MROWS * (CNH * CHD));         // attn out bf16
    (void)ws_size; (void)in_sizes; (void)n_in; (void)out_size;

    // 1) convert activations / transpose+convert weights
    {
        size_t n = MROWS * CH;
        cvt_f32_bf16<<<dim3((unsigned)((n + 255) / 256)), dim3(256), 0, stream>>>(x, xb, n);
    }
    transpose_cvt<<<dim3((unsigned)(((size_t)CH * 2048 + 255) / 256)), dim3(256), 0, stream>>>(wq, wqT, CH, 2048);
    transpose_cvt<<<dim3((unsigned)(((size_t)CH * 512  + 255) / 256)), dim3(256), 0, stream>>>(wk, wkT, CH, 512);
    transpose_cvt<<<dim3((unsigned)(((size_t)CH * 512  + 255) / 256)), dim3(256), 0, stream>>>(wv, wvT, CH, 512);
    transpose_cvt<<<dim3((unsigned)(((size_t)2048 * CH + 255) / 256)), dim3(256), 0, stream>>>(wo, woT, 2048, CH);

    // 2) QKV projections (bf16 WMMA GEMM, async-LDS double buffered)
    gemm_bf16_wmma<false><<<dim3(2048 / GBN, (unsigned)(MROWS / GBM)), dim3(256), 0, stream>>>(
        xb, wqT, nullptr, qraw, (int)MROWS, 2048, CH);
    gemm_bf16_wmma<false><<<dim3(512 / GBN, (unsigned)(MROWS / GBM)), dim3(256), 0, stream>>>(
        xb, wkT, nullptr, kraw, (int)MROWS, 512, CH);
    gemm_bf16_wmma<false><<<dim3(512 / GBN, (unsigned)(MROWS / GBM)), dim3(256), 0, stream>>>(
        xb, wvT, nullptr, vraw, (int)MROWS, 512, CH);

    // 3) RoPE + layout scatter
    rope_scatter<<<dim3((unsigned)((MROWS * CNH * CHD) / 256)), dim3(256), 0, stream>>>(
        qraw, cosb, sinb, qr, CNH);
    rope_scatter<<<dim3((unsigned)((MROWS * CNKV * CHD) / 256)), dim3(256), 0, stream>>>(
        kraw, cosb, sinb, kr, CNKV);
    v_scatter_t<<<dim3((unsigned)((MROWS * CNKV * CHD) / 256)), dim3(256), 0, stream>>>(vraw, vtb);

    // 4) causal flash attention (WMMA for QK^T and PV)
    flash_attn<<<dim3(CS / 16, CNH, CB), dim3(32), 0, stream>>>(qr, kr, vtb, ao);

    // 5) output projection -> f32 d_out
    gemm_bf16_wmma<true><<<dim3(2048 / GBN, (unsigned)(MROWS / GBM)), dim3(256), 0, stream>>>(
        ao, woT, out, nullptr, (int)MROWS, 2048, 2048);
}
